// MGEFE_8907762172241
// MI455X (gfx1250) — compile-verified, matrix operand order checked by reading, not run on hardware
//
#include <hip/hip_runtime.h>
#include <hip/hip_bf16.h>
#include <math.h>

typedef _Float16 f16;
typedef __attribute__((ext_vector_type(8)))  _Float16 v8h;
typedef __attribute__((ext_vector_type(16))) _Float16 v16h;
typedef __attribute__((ext_vector_type(8)))  float    v8f;

#define NPIX   65536      // 4*128*128
#define C      64
#define C3     192
#define HW     16384
#define HEADS  8

__device__ __forceinline__ float gelu_exact(float x) {
  return 0.5f * x * (1.0f + erff(x * 0.70710678118654752f));
}

// Load a 16-lane-half WMMA A/B fragment (f16, K=32 slice) from an LDS row.
// base points at the row (64 f16 stride assumed handled by caller), koff is the
// lane-dependent K offset: lane<16 -> {0..7,16..23}, lane>=16 -> {8..15,24..31}.
__device__ __forceinline__ v16h load_frag16(const f16* base, int koff) {
  const v8h lo = *reinterpret_cast<const v8h*>(base + koff);
  const v8h hi = *reinterpret_cast<const v8h*>(base + koff + 16);
  return __builtin_shufflevector(lo, hi, 0,1,2,3,4,5,6,7,8,9,10,11,12,13,14,15);
}

// ---------------------------------------------------------------------------
// Kernel 1: patch-embed (3 spectral convs + GELU) fused with GEMM1 (1536->64),
// epilogue (bias/affine/GELU/residual), and GEMM2 (qkv: 64->192). 128 px/block.
// ---------------------------------------------------------------------------
__global__ __launch_bounds__(256) void k1_embed_gemm(
    const float* __restrict__ x,
    const float* __restrict__ w3, const float* __restrict__ b3,
    const float* __restrict__ s3, const float* __restrict__ o3,
    const float* __restrict__ w5, const float* __restrict__ b5,
    const float* __restrict__ s5, const float* __restrict__ o5,
    const float* __restrict__ w7, const float* __restrict__ b7,
    const float* __restrict__ s7, const float* __restrict__ o7,
    const float* __restrict__ w2d, const float* __restrict__ b2d,
    const float* __restrict__ s2d, const float* __restrict__ o2d,
    const float* __restrict__ wqkv,
    f16* __restrict__ qkv_out)
{
  __shared__ __align__(16) f16 xs[128 * C];       // x tile (residual + conv src)
  __shared__ __align__(16) f16 ybuf[128 * C];     // conv slice / later ystage
  __shared__ __align__(16) f16 bstage[C3 * C];    // B panels (transposed)

  const int tid  = threadIdx.x;
  const int lane = tid & 31;
  const int wave = tid >> 5;
  const int p0   = blockIdx.x * 128;

  // stage x tile (f32 -> f16)
  for (int e = tid; e < 128 * C; e += 256) xs[e] = (f16)x[(size_t)p0 * C + e];
  __syncthreads();

  v8f c1[4] = {};
  const int rowbase = wave * 16;
  const f16* arow   = &ybuf[(rowbase + (lane & 15)) * C];
  const int koffb   = (lane & 16) ? 8 : 0;

  for (int j = 0; j < 24; ++j) {
    const int branch = j >> 3, o = j & 7;
    const float *wB, *bb, *ss, *oo; int K;
    if (branch == 0)      { wB = w3; bb = b3; ss = s3; oo = o3; K = 3; }
    else if (branch == 1) { wB = w5; bb = b5; ss = s5; oo = o5; K = 5; }
    else                  { wB = w7; bb = b7; ss = s7; oo = o7; K = 7; }
    float wt[7];
    for (int t = 0; t < K; ++t) wt[t] = wB[t * 8 + o];
    const float bj = bb[o], sj = ss[o], oj = oo[o];
    const int off = K >> 1;

    // stage B panel: w2dT[n][k] = w2d[(j*64+k)*64 + n]
    for (int e = tid; e < C * C; e += 256) {
      const int nn = e >> 6, kk = e & 63;
      bstage[nn * C + kk] = (f16)w2d[(size_t)(j * C + kk) * C + nn];
    }
    if (j < 23)
      __builtin_prefetch(&w2d[(size_t)((j + 1) * C) * C], 0, 1);

    // conv slice: ybuf[p][s] = gelu(s*(conv+b)+o)
    for (int e = tid; e < 128 * C; e += 256) {
      const int p = e >> 6, s = e & 63;
      float acc = 0.f;
      for (int t = 0; t < K; ++t) {
        const int idx = s + t - off;
        if (idx >= 0 && idx < C) acc += wt[t] * (float)xs[p * C + idx];
      }
      ybuf[e] = (f16)gelu_exact(sj * (acc + bj) + oj);
    }
    __syncthreads();

    // 16x64 x 64 WMMA accumulate (K=64 per group, 2 chunks of 32)
#pragma unroll
    for (int kc = 0; kc < 2; ++kc) {
      const v16h a = load_frag16(arow, kc * 32 + koffb);
#pragma unroll
      for (int nt = 0; nt < 4; ++nt) {
        const f16* bcol = &bstage[(nt * 16 + (lane & 15)) * C];
        const v16h bf = load_frag16(bcol, kc * 32 + koffb);
        c1[nt] = __builtin_amdgcn_wmma_f32_16x16x32_f16(
            false, a, false, bf, (short)0, c1[nt], false, false);
      }
    }
    __syncthreads();
  }

  // stage w_qkvT panel: [192][64]
  for (int e = tid; e < C3 * C; e += 256) {
    const int m = e >> 6, kk = e & 63;
    bstage[e] = (f16)wqkv[(size_t)kk * C3 + m];
  }

  // GEMM1 epilogue -> ystage (reuse ybuf; each wave touches only its own rows)
#pragma unroll
  for (int nt = 0; nt < 4; ++nt) {
#pragma unroll
    for (int r = 0; r < 8; ++r) {
      const int row = rowbase + r + ((lane & 16) ? 8 : 0);
      const int col = nt * 16 + (lane & 15);
      float v = c1[nt][r] + b2d[col];
      v = gelu_exact(v * s2d[col] + o2d[col]);
      v += (float)xs[row * C + col];        // residual
      ybuf[row * C + col] = (f16)v;
    }
  }
  __syncthreads();

  // GEMM2: qkv = y @ w_qkv  (K=64, N=192)
  v8f c2[12] = {};
#pragma unroll
  for (int kc = 0; kc < 2; ++kc) {
    const v16h a = load_frag16(arow, kc * 32 + koffb);
#pragma unroll
    for (int nt = 0; nt < 12; ++nt) {
      const f16* bcol = &bstage[(nt * 16 + (lane & 15)) * C];
      const v16h bf = load_frag16(bcol, kc * 32 + koffb);
      c2[nt] = __builtin_amdgcn_wmma_f32_16x16x32_f16(
          false, a, false, bf, (short)0, c2[nt], false, false);
    }
  }

#pragma unroll
  for (int nt = 0; nt < 12; ++nt) {
#pragma unroll
    for (int r = 0; r < 8; ++r) {
      const int row = rowbase + r + ((lane & 16) ? 8 : 0);
      const int col = nt * 16 + (lane & 15);
      qkv_out[(size_t)(p0 + row) * C3 + col] = (f16)c2[nt][r];
    }
  }
}

// ---------------------------------------------------------------------------
// Kernel 2: depthwise 3x3 SAME conv over HxW on 192 channels.
// ---------------------------------------------------------------------------
__global__ __launch_bounds__(256) void k2_dwconv(
    const f16* __restrict__ qkv, const float* __restrict__ wdw,
    float* __restrict__ qkvd)
{
  const int idx = blockIdx.x * 256 + threadIdx.x;
  if (idx >= NPIX * C3) return;
  const int m = idx % C3;
  const int p = idx / C3;
  const int b = p >> 14, rem = p & (HW - 1);
  const int y = rem >> 7, xq = rem & 127;
  float acc = 0.f;
#pragma unroll
  for (int dy = -1; dy <= 1; ++dy) {
    const int yy = y + dy;
    if (yy < 0 || yy >= 128) continue;
#pragma unroll
    for (int dx = -1; dx <= 1; ++dx) {
      const int xx = xq + dx;
      if (xx < 0 || xx >= 128) continue;
      const int pn = (b << 14) + (yy << 7) + xx;
      acc += (float)qkv[(size_t)pn * C3 + m] * wdw[((dy + 1) * 3 + (dx + 1)) * C3 + m];
    }
  }
  qkvd[(size_t)idx] = acc;
}

// ---------------------------------------------------------------------------
// Kernel 3: per (b,head) Gram(q,k) + |q|^2,|k|^2 over hw via LDS float
// atomics (ds_add_f32), then clipped-norm scaling + temperature + softmax.
// ---------------------------------------------------------------------------
__global__ __launch_bounds__(256) void k3_attn(
    const float* __restrict__ qkvd, const float* __restrict__ temperature,
    float* __restrict__ attnA)
{
  __shared__ float sg[80];    // [0..63]=Gqk, [64..71]=|q|^2, [72..79]=|k|^2
  __shared__ float satt[64];
  const int tid = threadIdx.x;
  const int bh = blockIdx.x, b = bh >> 3, head = bh & 7;
  if (tid < 80) sg[tid] = 0.f;
  __syncthreads();

  float aQK[64], aQ[8], aK[8], qv[8], kv[8];
#pragma unroll
  for (int i = 0; i < 64; ++i) aQK[i] = 0.f;
#pragma unroll
  for (int i = 0; i < 8; ++i) { aQ[i] = 0.f; aK[i] = 0.f; }

  for (int p = tid; p < HW; p += 256) {
    const float* base = qkvd + ((size_t)((b << 14) + p)) * C3 + head * 8;
#pragma unroll
    for (int i = 0; i < 8; ++i) { qv[i] = base[i]; kv[i] = base[64 + i]; }
#pragma unroll
    for (int ci = 0; ci < 8; ++ci)
#pragma unroll
      for (int d = 0; d < 8; ++d) aQK[ci * 8 + d] += qv[ci] * kv[d];
#pragma unroll
    for (int i = 0; i < 8; ++i) { aQ[i] += qv[i] * qv[i]; aK[i] += kv[i] * kv[i]; }
  }
#pragma unroll
  for (int i = 0; i < 64; ++i) atomicAdd(&sg[i], aQK[i]);
#pragma unroll
  for (int i = 0; i < 8; ++i) {
    atomicAdd(&sg[64 + i], aQ[i]);
    atomicAdd(&sg[72 + i], aK[i]);
  }
  __syncthreads();

  if (tid < 64) {
    const int ci = tid >> 3, d = tid & 7;
    const float nq = fmaxf(sqrtf(fmaxf(sg[64 + ci], 0.f)), 1e-12f);
    const float nk = fmaxf(sqrtf(fmaxf(sg[72 + d], 0.f)), 1e-12f);
    satt[tid] = sg[tid] / (nq * nk) * temperature[head];
  }
  __syncthreads();
  if (tid < 64) {
    const int ci = tid >> 3;
    const float* row = &satt[ci * 8];
    float m = row[0];
#pragma unroll
    for (int d = 1; d < 8; ++d) m = fmaxf(m, row[d]);
    float s = 0.f;
#pragma unroll
    for (int d = 0; d < 8; ++d) s += expf(row[d] - m);
    attnA[bh * 64 + tid] = expf(satt[tid] - m) / s;
  }
}

// ---------------------------------------------------------------------------
// Kernel 5: out = (attn @ v per pixel/head) then @ w_proj via WMMA.
// ---------------------------------------------------------------------------
__global__ __launch_bounds__(256) void k5_mix_proj(
    const float* __restrict__ qkvd, const float* __restrict__ attnA,
    const float* __restrict__ wproj, float* __restrict__ out)
{
  __shared__ __align__(16) f16 wpT[C * C];     // [cp][j]
  __shared__ __align__(16) f16 ost[128 * C];   // mixed features, f16
  __shared__ float aA[512];                    // attn for this batch [head][c][d]

  const int tid  = threadIdx.x;
  const int lane = tid & 31;
  const int wave = tid >> 5;
  const int p0   = blockIdx.x * 128;
  const int b    = p0 >> 14;

  for (int e = tid; e < C * C; e += 256) {
    const int cp = e >> 6, j = e & 63;
    wpT[e] = (f16)wproj[(size_t)j * C + cp];
  }
  for (int e = tid; e < 512; e += 256) aA[e] = attnA[b * 512 + e];
  __syncthreads();

  for (int e = tid; e < 128 * C; e += 256) {
    const int pl = e >> 6, jc = e & 63;
    const int head = jc >> 3, cd = jc & 7;
    const float* vb = qkvd + (size_t)(p0 + pl) * C3 + 128 + head * 8;
    float acc = 0.f;
#pragma unroll
    for (int d = 0; d < 8; ++d) acc += aA[head * 64 + cd * 8 + d] * vb[d];
    ost[e] = (f16)acc;
  }
  __syncthreads();

  const int rowbase = wave * 16;
  const f16* arow   = &ost[(rowbase + (lane & 15)) * C];
  const int koffb   = (lane & 16) ? 8 : 0;
  v8f cc[4] = {};
#pragma unroll
  for (int kc = 0; kc < 2; ++kc) {
    const v16h a = load_frag16(arow, kc * 32 + koffb);
#pragma unroll
    for (int nt = 0; nt < 4; ++nt) {
      const f16* bcol = &wpT[(nt * 16 + (lane & 15)) * C];
      const v16h bf = load_frag16(bcol, kc * 32 + koffb);
      cc[nt] = __builtin_amdgcn_wmma_f32_16x16x32_f16(
          false, a, false, bf, (short)0, cc[nt], false, false);
    }
  }
#pragma unroll
  for (int nt = 0; nt < 4; ++nt) {
#pragma unroll
    for (int r = 0; r < 8; ++r) {
      const int row = rowbase + r + ((lane & 16) ? 8 : 0);
      const int col = nt * 16 + (lane & 15);
      out[(size_t)(p0 + row) * C + col] = cc[nt][r];
    }
  }
}

// ---------------------------------------------------------------------------
extern "C" void kernel_launch(void* const* d_in, const int* in_sizes, int n_in,
                              void* d_out, int out_size, void* d_ws, size_t ws_size,
                              hipStream_t stream) {
  const float* x    = (const float*)d_in[0];
  const float* w3   = (const float*)d_in[1];
  const float* b3   = (const float*)d_in[2];
  const float* s3   = (const float*)d_in[3];
  const float* o3   = (const float*)d_in[4];
  const float* w5   = (const float*)d_in[5];
  const float* b5   = (const float*)d_in[6];
  const float* s5   = (const float*)d_in[7];
  const float* o5   = (const float*)d_in[8];
  const float* w7   = (const float*)d_in[9];
  const float* b7   = (const float*)d_in[10];
  const float* s7   = (const float*)d_in[11];
  const float* o7   = (const float*)d_in[12];
  const float* w2d  = (const float*)d_in[13];
  const float* b2d  = (const float*)d_in[14];
  const float* s2d  = (const float*)d_in[15];
  const float* o2d  = (const float*)d_in[16];
  const float* wqkv = (const float*)d_in[17];
  const float* wdw  = (const float*)d_in[18];
  const float* wprj = (const float*)d_in[19];
  const float* temp = (const float*)d_in[20];

  // workspace layout: qkv (f16, 24MB) | qkvd (f32, 48MB) | attnA (f32, 8KB)
  char* ws = (char*)d_ws;
  f16*   qkv   = (f16*)ws;
  float* qkvd  = (float*)(ws + (size_t)NPIX * C3 * sizeof(f16));
  float* attnA = (float*)(ws + (size_t)NPIX * C3 * (sizeof(f16) + sizeof(float)));
  float* out   = (float*)d_out;

  k1_embed_gemm<<<NPIX / 128, 256, 0, stream>>>(
      x, w3, b3, s3, o3, w5, b5, s5, o5, w7, b7, s7, o7,
      w2d, b2d, s2d, o2d, wqkv, qkv);
  k2_dwconv<<<(NPIX * C3) / 256, 256, 0, stream>>>(qkv, wdw, qkvd);
  k3_attn<<<4 * HEADS, 256, 0, stream>>>(qkvd, temp, attnA);
  k5_mix_proj<<<NPIX / 128, 256, 0, stream>>>(qkvd, attnA, wprj, out);
}
